// MaskedMultiHeadAttention_43069932045110
// MI455X (gfx1250) — compile-verified, matrix operand order checked by reading, not run on hardware
//
#include <hip/hip_runtime.h>

// ---------------------------------------------------------------------------
// Masked multi-head attention with RoPE for MI455X (gfx1250, wave32, WMMA).
// Compute-bound (~103 GFLOP vs ~60MB of data that is L2-resident on a 192MB
// L2) -> all matmuls run on v_wmma_f32_16x16x32_bf16 with f32 accumulation.
// GEMM waves compute 32x64 tiles (8 wmma per k-step, B-fragments reused 2x);
// attention is flash-style with online softmax; global_prefetch_b8 hides
// latency for the single-wave workgroups.
// ---------------------------------------------------------------------------

#define D_MODEL   1024
#define NUM_HEADS 16
#define HEAD_DIM  64
#define BATCH     2
#define SEQ       2048
#define NTOK      (BATCH * SEQ)

typedef __bf16 bf16;
typedef __bf16 v16bf __attribute__((ext_vector_type(16)));
typedef __bf16 v8bf  __attribute__((ext_vector_type(8)));
typedef __bf16 v4bf  __attribute__((ext_vector_type(4)));
typedef float  v8f   __attribute__((ext_vector_type(8)));

__device__ __forceinline__ bf16 f2bf(float f) {
    unsigned u = __builtin_bit_cast(unsigned, f);
    u += 0x7FFFu + ((u >> 16) & 1u);               // round-to-nearest-even
    unsigned short h = (unsigned short)(u >> 16);
    return __builtin_bit_cast(bf16, h);
}

// Load a 16x32 bf16 fragment in the wave32 A/B-operand layout:
//   lane L (L<16):  row = L,    K offsets {k0+0..7, k0+16..23}
//   lane L (L>=16): row = L-16, K offsets {k0+8..15, k0+24..31}
// Works for A (rows = M of x/Q/P) and B (rows = N: row n of W / K / Vt is
// column n of the B matrix, contiguous along K).
__device__ __forceinline__ v16bf load_frag(const bf16* __restrict__ p, int ld,
                                           int row0, int k0) {
    const int lane = threadIdx.x & 31;
    const int r    = lane & 15;
    const int kk   = k0 + ((lane >> 4) << 3);
    const bf16* rp = p + (size_t)(row0 + r) * (size_t)ld + kk;
    v8bf lo = *(const v8bf*)(rp);
    v8bf hi = *(const v8bf*)(rp + 16);
    v16bf out;
#pragma unroll
    for (int i = 0; i < 8; ++i) { out[i] = lo[i]; out[i + 8] = hi[i]; }
    return out;
}

__device__ __forceinline__ v8f wmma_bf16(v16bf a, v16bf b, v8f c) {
    return __builtin_amdgcn_wmma_f32_16x16x32_bf16(
        /*neg_a=*/false, a, /*neg_b=*/false, b,
        /*c_mod=*/(short)0, c, /*reuse_a=*/false, /*reuse_b=*/false);
}

// ---------------------------------------------------------------------------
__global__ void cvt_f32_to_bf16(const float4* __restrict__ in,
                                v4bf* __restrict__ out, int n4) {
    int i = blockIdx.x * blockDim.x + threadIdx.x;
    const int stride = gridDim.x * blockDim.x;
    for (; i < n4; i += stride) {
        float4 f = in[i];
        v4bf o;
        o[0] = f2bf(f.x); o[1] = f2bf(f.y); o[2] = f2bf(f.z); o[3] = f2bf(f.w);
        out[i] = o;
    }
}

// ---------------------------------------------------------------------------
// QKV projection: one wave = 32 tokens x 64 features (one head).
// mode 0: +bias, RoPE, *1/sqrt(Dh), store [B,H,S,Dh]   (Q, pre-scaled)
// mode 1: +bias, RoPE,              store [B,H,S,Dh]   (K)
// mode 2: +bias,                    store Vt [B,H,Dh,S] (V transposed)
__global__ void __launch_bounds__(32)
qkv_proj_kernel(const bf16* __restrict__ xb, const bf16* __restrict__ Wb,
                const float* __restrict__ bias, bf16* __restrict__ outp,
                int mode) {
    const int t0   = blockIdx.x * 32;     // token tile base
    const int h    = blockIdx.y;          // head
    const int n0   = h * HEAD_DIM;        // output-feature base
    const int lane = threadIdx.x & 31;
    const int c    = lane & 15;
    const int half = lane >> 4;

    v8f acc[2][4];
#pragma unroll
    for (int g = 0; g < 2; ++g)
#pragma unroll
        for (int t = 0; t < 4; ++t)
#pragma unroll
            for (int i = 0; i < 8; ++i) acc[g][t][i] = 0.0f;

    for (int k0 = 0; k0 < D_MODEL; k0 += 32) {
        const int kn = k0 + 64;
        if (kn < D_MODEL) {   // prefetch two k-steps ahead (global_prefetch_b8)
            __builtin_prefetch(xb + (size_t)(t0 + lane) * D_MODEL + kn, 0, 1);
            __builtin_prefetch(Wb + (size_t)(n0 + 2 * lane) * D_MODEL + kn, 0, 1);
        }
        v16bf a0 = load_frag(xb, D_MODEL, t0, k0);
        v16bf a1 = load_frag(xb, D_MODEL, t0 + 16, k0);
#pragma unroll
        for (int t = 0; t < 4; ++t) {
            v16bf b = load_frag(Wb, D_MODEL, n0 + t * 16, k0);
            acc[0][t] = wmma_bf16(a0, b, acc[0][t]);
            acc[1][t] = wmma_bf16(a1, b, acc[1][t]);
        }
    }

    float bvals[4];
#pragma unroll
    for (int t = 0; t < 4; ++t) bvals[t] = bias[n0 + t * 16 + c];

#pragma unroll
    for (int g = 0; g < 2; ++g) {
#pragma unroll
        for (int i = 0; i < 8; ++i) {
            const int tok   = t0 + g * 16 + i + 8 * half;  // C row = i + 8*half
            const int b_idx = tok / SEQ;
            const int pos   = tok - b_idx * SEQ;
            float v[4];
#pragma unroll
            for (int t = 0; t < 4; ++t) v[t] = acc[g][t][i] + bvals[t];

            if (mode < 2) {
                // rotate-half RoPE: pair d (tile t) <-> d+32 (tile t+2)
#pragma unroll
                for (int t = 0; t < 2; ++t) {
                    const int j = t * 16 + c;                   // 0..31
                    float inv = __expf((float)(2 * j) * (-9.21034037198f / 64.0f));
                    float ang = (float)pos * inv;
                    float cs = __cosf(ang), sn = __sinf(ang);
                    float x1 = v[t], x2 = v[t + 2];
                    v[t]     = x1 * cs - x2 * sn;
                    v[t + 2] = x2 * cs + x1 * sn;
                }
                if (mode == 0) {            // fold 1/sqrt(Dh) into Q
#pragma unroll
                    for (int t = 0; t < 4; ++t) v[t] *= 0.125f;
                }
                bf16* o = outp +
                    ((size_t)(b_idx * NUM_HEADS + h) * SEQ + pos) * HEAD_DIM;
#pragma unroll
                for (int t = 0; t < 4; ++t) o[t * 16 + c] = f2bf(v[t]);
            } else {
                bf16* o = outp + (size_t)(b_idx * NUM_HEADS + h) * HEAD_DIM * SEQ;
#pragma unroll
                for (int t = 0; t < 4; ++t)
                    o[(size_t)(t * 16 + c) * SEQ + pos] = f2bf(v[t]);
            }
        }
    }
}

// ---------------------------------------------------------------------------
// Flash attention: one wave per (b,h, 16-query tile); 32-key blocks.
// Q arrives pre-scaled by 1/sqrt(Dh).
__global__ void __launch_bounds__(32)
attn_kernel(const bf16* __restrict__ Q, const bf16* __restrict__ K,
            const bf16* __restrict__ Vt, bf16* __restrict__ ctx) {
    __shared__ alignas(16) bf16 pbuf[16 * 32];  // P tile: C-layout -> A-layout

    const int qb    = blockIdx.x * 16;
    const int bh    = blockIdx.y;
    const int b_idx = bh >> 4;
    const int h     = bh & 15;
    const int lane  = threadIdx.x & 31;
    const int c     = lane & 15;
    const int half  = lane >> 4;

    const bf16* Qh = Q  + (size_t)bh * SEQ * HEAD_DIM;
    const bf16* Kh = K  + (size_t)bh * SEQ * HEAD_DIM;
    const bf16* Vh = Vt + (size_t)bh * HEAD_DIM * SEQ;

    v16bf aq[2];
    aq[0] = load_frag(Qh, HEAD_DIM, qb, 0);
    aq[1] = load_frag(Qh, HEAD_DIM, qb, 32);

    v8f o[4];
    float mrow[8], lrow[8];
#pragma unroll
    for (int t = 0; t < 4; ++t)
#pragma unroll
        for (int i = 0; i < 8; ++i) o[t][i] = 0.0f;
#pragma unroll
    for (int i = 0; i < 8; ++i) { mrow[i] = -1e30f; lrow[i] = 0.0f; }

    for (int kb = 0; kb < qb + 16; kb += 32) {
        const int nk = kb + 32;
        if (nk < qb + 16) {   // prefetch next k-block's K rows / Vt rows
            __builtin_prefetch(Kh + (size_t)(nk + lane) * HEAD_DIM, 0, 1);
            __builtin_prefetch(Vh + (size_t)(2 * lane) * SEQ + nk, 0, 1);
        }

        // ---- scores for 32 keys (2 x 16-col tiles); Q pre-scaled
        v8f s[2];
#pragma unroll
        for (int kt = 0; kt < 2; ++kt) {
#pragma unroll
            for (int i = 0; i < 8; ++i) s[kt][i] = 0.0f;
#pragma unroll
            for (int j = 0; j < 2; ++j) {
                v16bf bk = load_frag(Kh, HEAD_DIM, kb + kt * 16, j * 32);
                s[kt] = wmma_bf16(aq[j], bk, s[kt]);
            }
        }

        // ---- causal mask + row max (rows live in 16-lane halves)
        float rmax[8];
#pragma unroll
        for (int i = 0; i < 8; ++i) {
            const int qg = qb + i + 8 * half;
            float a0 = s[0][i];
            float a1 = s[1][i];
            if (kb + c > qg)      a0 = -1e30f;
            if (kb + 16 + c > qg) a1 = -1e30f;
            s[0][i] = a0; s[1][i] = a1;
            float r = fmaxf(a0, a1);
            r = fmaxf(r, __shfl_xor(r, 1, 32));
            r = fmaxf(r, __shfl_xor(r, 2, 32));
            r = fmaxf(r, __shfl_xor(r, 4, 32));
            r = fmaxf(r, __shfl_xor(r, 8, 32));
            rmax[i] = r;
        }

        // ---- online softmax update
        float alpha[8];
#pragma unroll
        for (int i = 0; i < 8; ++i) {
            float mn = fmaxf(mrow[i], rmax[i]);
            alpha[i] = __expf(mrow[i] - mn);
            mrow[i]  = mn;
        }

        float rsum[8];
#pragma unroll
        for (int i = 0; i < 8; ++i) {
            float p0 = __expf(s[0][i] - mrow[i]);
            float p1 = __expf(s[1][i] - mrow[i]);
            pbuf[(i + 8 * half) * 32 + c]      = f2bf(p0);
            pbuf[(i + 8 * half) * 32 + 16 + c] = f2bf(p1);
            float rs = p0 + p1;
            rs += __shfl_xor(rs, 1, 32);
            rs += __shfl_xor(rs, 2, 32);
            rs += __shfl_xor(rs, 4, 32);
            rs += __shfl_xor(rs, 8, 32);
            rsum[i] = rs;
        }
#pragma unroll
        for (int i = 0; i < 8; ++i) lrow[i] = lrow[i] * alpha[i] + rsum[i];
#pragma unroll
        for (int t = 0; t < 4; ++t)
#pragma unroll
            for (int i = 0; i < 8; ++i) o[t][i] *= alpha[i];

        __syncthreads();
        // ---- reload P as an A fragment (two 16B ds_load_b128 per lane)
        v16bf ap;
        {
            const bf16* pr = &pbuf[c * 32 + half * 8];
            v8bf lo = *(const v8bf*)(pr);
            v8bf hi = *(const v8bf*)(pr + 16);
#pragma unroll
            for (int i = 0; i < 8; ++i) { ap[i] = lo[i]; ap[i + 8] = hi[i]; }
        }
        __syncthreads();

        // ---- O += P @ V   (V transposed: contiguous along keys)
#pragma unroll
        for (int t = 0; t < 4; ++t) {
            v16bf bv = load_frag(Vh, SEQ, t * 16, kb);
            o[t] = wmma_bf16(ap, bv, o[t]);
        }
    }

    // ---- finalize: O / l -> ctx[token][h*64 + d] (bf16)
#pragma unroll
    for (int i = 0; i < 8; ++i) {
        const float inv = 1.0f / lrow[i];
        const int tok = b_idx * SEQ + qb + i + 8 * half;
        bf16* op = ctx + (size_t)tok * D_MODEL + h * HEAD_DIM;
#pragma unroll
        for (int t = 0; t < 4; ++t) op[t * 16 + c] = f2bf(o[t][i] * inv);
    }
}

// ---------------------------------------------------------------------------
// Output projection: one wave = 32 tokens x 64 features; out = ctx @ Wo^T + bo
__global__ void __launch_bounds__(32)
out_proj_kernel(const bf16* __restrict__ ctx, const bf16* __restrict__ Wob,
                const float* __restrict__ bo, float* __restrict__ out) {
    const int t0   = blockIdx.x * 32;
    const int n0   = blockIdx.y * 64;
    const int lane = threadIdx.x & 31;
    const int c    = lane & 15;
    const int half = lane >> 4;

    v8f acc[2][4];
#pragma unroll
    for (int g = 0; g < 2; ++g)
#pragma unroll
        for (int t = 0; t < 4; ++t)
#pragma unroll
            for (int i = 0; i < 8; ++i) acc[g][t][i] = 0.0f;

    for (int k0 = 0; k0 < D_MODEL; k0 += 32) {
        const int kn = k0 + 64;
        if (kn < D_MODEL) {
            __builtin_prefetch(ctx + (size_t)(t0 + lane) * D_MODEL + kn, 0, 1);
            __builtin_prefetch(Wob + (size_t)(n0 + 2 * lane) * D_MODEL + kn, 0, 1);
        }
        v16bf a0 = load_frag(ctx, D_MODEL, t0, k0);
        v16bf a1 = load_frag(ctx, D_MODEL, t0 + 16, k0);
#pragma unroll
        for (int t = 0; t < 4; ++t) {
            v16bf b = load_frag(Wob, D_MODEL, n0 + t * 16, k0);
            acc[0][t] = wmma_bf16(a0, b, acc[0][t]);
            acc[1][t] = wmma_bf16(a1, b, acc[1][t]);
        }
    }

#pragma unroll
    for (int g = 0; g < 2; ++g) {
#pragma unroll
        for (int i = 0; i < 8; ++i) {
            const int tok = t0 + g * 16 + i + 8 * half;
            float* op = out + (size_t)tok * D_MODEL;
#pragma unroll
            for (int t = 0; t < 4; ++t)
                op[n0 + t * 16 + c] = acc[g][t][i] + bo[n0 + t * 16 + c];
        }
    }
}

// ---------------------------------------------------------------------------
extern "C" void kernel_launch(void* const* d_in, const int* in_sizes, int n_in,
                              void* d_out, int out_size, void* d_ws, size_t ws_size,
                              hipStream_t stream) {
    const float* x  = (const float*)d_in[0];
    const float* Wq = (const float*)d_in[1];
    const float* bq = (const float*)d_in[2];
    const float* Wk = (const float*)d_in[3];
    const float* bk = (const float*)d_in[4];
    const float* Wv = (const float*)d_in[5];
    const float* bv = (const float*)d_in[6];
    const float* Wo = (const float*)d_in[7];
    const float* bo = (const float*)d_in[8];
    float* out = (float*)d_out;

    char* ws = (char*)d_ws;
    const size_t MiB = (size_t)1 << 20;
    // Regions (Q/K/Vt get +1MiB slack for masked over-reads at causal edge):
    bf16* xb  = (bf16*)(ws + 0 * MiB);   // 8 MiB  x in bf16 [4096,1024]
    bf16* Wqb = (bf16*)(ws + 8 * MiB);   // 2 MiB
    bf16* Wkb = (bf16*)(ws + 10 * MiB);  // 2 MiB
    bf16* Wvb = (bf16*)(ws + 12 * MiB);  // 2 MiB
    bf16* Wob = (bf16*)(ws + 14 * MiB);  // 2 MiB
    bf16* Qb  = (bf16*)(ws + 16 * MiB);  // 8+1 MiB [B,H,S,Dh]
    bf16* Kb  = (bf16*)(ws + 25 * MiB);  // 8+1 MiB [B,H,S,Dh]
    bf16* Vtb = (bf16*)(ws + 34 * MiB);  // 8+1 MiB [B,H,Dh,S]
    bf16* ctx = (bf16*)(ws + 43 * MiB);  // 8 MiB  [4096,1024]

    cvt_f32_to_bf16<<<512, 256, 0, stream>>>((const float4*)x, (v4bf*)xb,
                                             NTOK * D_MODEL / 4);
    cvt_f32_to_bf16<<<256, 256, 0, stream>>>((const float4*)Wq, (v4bf*)Wqb,
                                             D_MODEL * D_MODEL / 4);
    cvt_f32_to_bf16<<<256, 256, 0, stream>>>((const float4*)Wk, (v4bf*)Wkb,
                                             D_MODEL * D_MODEL / 4);
    cvt_f32_to_bf16<<<256, 256, 0, stream>>>((const float4*)Wv, (v4bf*)Wvb,
                                             D_MODEL * D_MODEL / 4);
    cvt_f32_to_bf16<<<256, 256, 0, stream>>>((const float4*)Wo, (v4bf*)Wob,
                                             D_MODEL * D_MODEL / 4);

    dim3 gq(NTOK / 32, NUM_HEADS);
    qkv_proj_kernel<<<gq, 32, 0, stream>>>(xb, Wqb, bq, Qb,  0);
    qkv_proj_kernel<<<gq, 32, 0, stream>>>(xb, Wkb, bk, Kb,  1);
    qkv_proj_kernel<<<gq, 32, 0, stream>>>(xb, Wvb, bv, Vtb, 2);

    dim3 ga(SEQ / 16, BATCH * NUM_HEADS);
    attn_kernel<<<ga, 32, 0, stream>>>(Qb, Kb, Vtb, ctx);

    dim3 go(NTOK / 32, D_MODEL / 64);
    out_proj_kernel<<<go, 32, 0, stream>>>(ctx, Wob, bo, out);
}
